// Kim2017_9672266351068
// MI455X (gfx1250) — matrix operand order repaired and checked by reading, not compiled, not run on hardware
//
#include <hip/hip_runtime.h>

// ---------------- types for CDNA5 WMMA ----------------
typedef __bf16 bf16_t;
typedef __attribute__((ext_vector_type(16))) __bf16 v16bf;
typedef __attribute__((ext_vector_type(8)))  __bf16 v8bf;
typedef __attribute__((ext_vector_type(8)))  float  v8f;

// Hard-coded dims from the reference
#define DIM_D 64
#define DIM_L 90
#define DIM_R 5760
#define DIM_H 1024
#define DIM_T 20
#define DIM_B 64
#define KCAT  (DIM_R + DIM_H)   // 6784
#define NSPLIT 4                // split-K factor for the two big GEMMs

static __device__ __forceinline__ v16bf cat8(v8bf lo, v8bf hi) {
  return __builtin_shufflevector(lo, hi, 0,1,2,3,4,5,6,7,8,9,10,11,12,13,14,15);
}

#define WMMA_BF16(ACC, AF, BF)                                            \
  (ACC) = __builtin_amdgcn_wmma_f32_16x16x32_bf16(false, (AF), false,     \
                                                  (BF), (short)0, (ACC),  \
                                                  false, false)

// ---------------------------------------------------------------------------
// bf16 WMMA GEMM with split-K: C[z][M,N] = A[M,Kz] * W[N,Kz]^T  (z = blockIdx.z)
// Block: 256 threads = 8 waves; wave owns a 16-wide N strip x 4 stacked M tiles.
// Grid: (ceil(N/128), M/64, nsplit).
// k-loop: explicit ping-pong over chunk PAIRS with two live fragment sets, so
// the register allocator cannot coalesce the double buffer; each 4-WMMA group
// runs with the other buffer's 10 global_load_b128 in flight (partial waits).
// featmap!=0 remaps A rows for the [B,T,L,D]->[T,B,L,D] feature transpose.
// ---------------------------------------------------------------------------
__global__ void __launch_bounds__(256)
gemm_wmma_bf16(const bf16_t* __restrict__ A, int lda,
               const bf16_t* __restrict__ W, int ldw,
               float* __restrict__ C, int ldc,
               int M, int N, int K, int featmap, int t)
{
  const int lane = threadIdx.x & 31;
  const int wave = threadIdx.x >> 5;
  const int lmod = lane & 15;
  const int lhi  = lane >> 4;            // 0: lanes 0-15, 1: lanes 16-31
  const int mb   = blockIdx.y;
  const int z    = blockIdx.z;
  const int n    = blockIdx.x * 128 + wave * 16 + lmod;
  const int n_eff = (n < N) ? n : (N - 1);

  // split-K range: per-split chunk count rounded up to EVEN (pair loop)
  const int kchunks = K >> 5;
  int per = (kchunks + (int)gridDim.z - 1) / (int)gridDim.z;
  per = (per + 1) & ~1;
  const int ks = z * per * 32;
  int ke = ks + per * 32;
  if (ke > K) ke = K;

  float* Cz = C + (size_t)z * M * ldc;

  if (ks >= ke) {                        // empty split: write zeros (reducers
    if (n < N) {                         // sum all slabs unconditionally)
#pragma unroll
      for (int mt = 0; mt < 4; ++mt) {
        const int rbase = mb * 64 + mt * 16 + 8 * lhi;
#pragma unroll
        for (int e = 0; e < 8; ++e) {
          int row = rbase + e;
          if (row < M) Cz[(size_t)row * ldc + n] = 0.0f;
        }
      }
    }
    return;
  }

  // B fragment (K x 16): lane holds W row n, K +0..7/+16..23 (lhi=0) or +8..15/+24..31 (lhi=1)
  const bf16_t* Wbase = W + (size_t)n_eff * ldw + lhi * 8;

  // A fragments: lane holds row m = lmod of each 16-row tile
  const bf16_t* Ab0;
  const bf16_t* Ab1;
  const bf16_t* Ab2;
  const bf16_t* Ab3;
  {
    const bf16_t* tmp[4];
#pragma unroll
    for (int mt = 0; mt < 4; ++mt) {
      int r  = mb * 64 + mt * 16 + lmod;
      int sr = (r < M) ? r : (M - 1);
      if (featmap) {                     // r = b*L + l  ->  (b*T + t)*L + l
        int b = sr / DIM_L;
        int l = sr - b * DIM_L;
        sr = (b * DIM_T + t) * DIM_L + l;
      }
      tmp[mt] = A + (size_t)sr * lda + lhi * 8;
    }
    Ab0 = tmp[0]; Ab1 = tmp[1]; Ab2 = tmp[2]; Ab3 = tmp[3];
  }

  auto loadB = [&](int k0) -> v16bf {
    const v8bf* wp = (const v8bf*)(Wbase + k0);          // +0B, +32B
    return cat8(wp[0], wp[2]);
  };
  auto loadA = [&](const bf16_t* base, int k0) -> v16bf {
    const v8bf* p0 = (const v8bf*)(base + k0);           // K +0..7 (or +8..15)
    const v8bf* p1 = (const v8bf*)(base + k0 + 16);      // K +16..23 (or +24..31)
    return cat8(*p0, *p1);
  };

  v8f acc0 = {}, acc1 = {}, acc2 = {}, acc3 = {};

  // prologue: fill buffer P with chunk ks
  v16bf pB  = loadB(ks);
  v16bf pA0 = loadA(Ab0, ks);
  v16bf pA1 = loadA(Ab1, ks);
  v16bf pA2 = loadA(Ab2, ks);
  v16bf pA3 = loadA(Ab3, ks);

  int k0 = ks;
  for (;;) {
    const int k1 = k0 + 32;              // exists: even chunk count per split
    // issue buffer-Q loads, then compute on P (Q loads stay in flight)
    v16bf qB  = loadB(k1);
    v16bf qA0 = loadA(Ab0, k1);
    v16bf qA1 = loadA(Ab1, k1);
    v16bf qA2 = loadA(Ab2, k1);
    v16bf qA3 = loadA(Ab3, k1);

    WMMA_BF16(acc0, pA0, pB);
    WMMA_BF16(acc1, pA1, pB);
    WMMA_BF16(acc2, pA2, pB);
    WMMA_BF16(acc3, pA3, pB);

    const int k2 = k1 + 32;
    const bool more = (k2 < ke);
    const int kp = more ? k2 : ks;       // clamped dangling prefetch on exit
    // refill P, then compute on Q (P loads stay in flight)
    pB  = loadB(kp);
    pA0 = loadA(Ab0, kp);
    pA1 = loadA(Ab1, kp);
    pA2 = loadA(Ab2, kp);
    pA3 = loadA(Ab3, kp);

    WMMA_BF16(acc0, qA0, qB);
    WMMA_BF16(acc1, qA1, qB);
    WMMA_BF16(acc2, qA2, qB);
    WMMA_BF16(acc3, qA3, qB);

    if (!more) break;
    k0 = k2;
  }

  if (n < N) {
    v8f accs[4] = {acc0, acc1, acc2, acc3};
#pragma unroll
    for (int mt = 0; mt < 4; ++mt) {
      const int rbase = mb * 64 + mt * 16 + 8 * lhi;     // C/D: VGPR e -> M=e(+8)
#pragma unroll
      for (int e = 0; e < 8; ++e) {
        int row = rbase + e;
        if (row < M) Cz[(size_t)row * ldc + n] = accs[mt][e];
      }
    }
  }
}

// ---------------------------------------------------------------------------
// One-time weight prep kernels
// ---------------------------------------------------------------------------
__global__ void build_cat_bf16(const float* __restrict__ Wa,
                               const float* __restrict__ Wb,
                               bf16_t* __restrict__ dst,
                               int N, int Ka, int Kb)
{
  const long long Kc = Ka + Kb;
  long long i = (long long)blockIdx.x * 256 + threadIdx.x;
  if (i >= (long long)N * Kc) return;
  int n = (int)(i / Kc);
  int k = (int)(i - (long long)n * Kc);
  float v = (k < Ka) ? Wa[(size_t)n * Ka + k] : Wb[(size_t)n * Kb + (k - Ka)];
  dst[i] = (bf16_t)v;
}

__global__ void cvt_bf16(const float* __restrict__ src, bf16_t* __restrict__ dst,
                         long long n)
{
  long long i = (long long)blockIdx.x * 256 + threadIdx.x;
  if (i < n) dst[i] = (bf16_t)src[i];
}

__global__ void misc_init(const float* __restrict__ attn_Wx,
                          const float* __restrict__ b_ih,
                          const float* __restrict__ b_hh,
                          bf16_t* __restrict__ WxT,
                          float* __restrict__ bias_g)
{
  int i = blockIdx.x * 256 + threadIdx.x;
  if (i < DIM_D * DIM_D) {
    int a = i >> 6, d = i & 63;
    WxT[a * DIM_D + d] = (bf16_t)attn_Wx[d * DIM_D + a];
  }
  if (i < 4 * DIM_H) bias_g[i] = b_ih[i] + b_hh[i];
}

// ---------------------------------------------------------------------------
// h0/c0 init; also seeds the h-slice of the bf16 activation buffer Acat.
// ---------------------------------------------------------------------------
__global__ void __launch_bounds__(256)
init_state(const float* __restrict__ feature,
           const float* __restrict__ iWh_w, const float* __restrict__ iWh_b,
           const float* __restrict__ iWc_w, const float* __restrict__ iWc_b,
           float* __restrict__ h, float* __restrict__ c,
           bf16_t* __restrict__ Acat)
{
  const int b = blockIdx.x;
  __shared__ float f0[DIM_D];
  if (threadIdx.x < DIM_D) {
    int d = threadIdx.x;
    const float* base = feature + (size_t)(b * DIM_T) * (DIM_L * DIM_D) + d;
    float s = 0.f;
    for (int l = 0; l < DIM_L; ++l) s += base[l * DIM_D];
    f0[d] = s * (1.0f / DIM_L);
  }
  __syncthreads();
  for (int j = threadIdx.x; j < DIM_H; j += 256) {
    const float* wr = iWh_w + (size_t)j * DIM_D;
    const float* wc = iWc_w + (size_t)j * DIM_D;
    float sh = iWh_b[j], sc = iWc_b[j];
    for (int d = 0; d < DIM_D; ++d) { sh += f0[d] * wr[d]; sc += f0[d] * wc[d]; }
    float hv = tanhf(sh), cv = tanhf(sc);
    h[b * DIM_H + j] = hv;
    c[b * DIM_H + j] = cv;
    Acat[(size_t)b * KCAT + DIM_R + j] = (bf16_t)hv;
  }
}

// ---------------------------------------------------------------------------
// Attention: e = tanh(xW + h@Wh) . v ; softmax over L ; visual -> Acat (bf16)
// One block per batch row b. h@Wh uses all 256 threads (4-way K partials).
// ---------------------------------------------------------------------------
__global__ void __launch_bounds__(256)
attn_kernel(const float* __restrict__ feature,
            const float* __restrict__ attn_Wh,   // [H, D] row-major
            const float* __restrict__ attn_v,    // [D]
            const float* __restrict__ xW,        // [B*L, D] fp32
            const float* __restrict__ h,         // [B, H] fp32 state
            bf16_t* __restrict__ Acat, int t)
{
  const int b = blockIdx.x;
  __shared__ float sh[DIM_H];
  __shared__ float hWp[4][DIM_D];
  __shared__ float hW[DIM_D];
  __shared__ float sE[DIM_L];
  __shared__ float sA[DIM_L];

  for (int j = threadIdx.x; j < DIM_H; j += 256) sh[j] = h[b * DIM_H + j];
  __syncthreads();

  {
    int a = threadIdx.x & 63;
    int q = threadIdx.x >> 6;                // 4 quarters of K=1024
    float s = 0.f;
    for (int j = q * 256; j < (q + 1) * 256; ++j)
      s += sh[j] * attn_Wh[(size_t)j * DIM_D + a];
    hWp[q][a] = s;
  }
  __syncthreads();
  if (threadIdx.x < DIM_D) {
    int a = threadIdx.x;
    hW[a] = hWp[0][a] + hWp[1][a] + hWp[2][a] + hWp[3][a];
  }
  __syncthreads();

  if (threadIdx.x < DIM_L) {
    int l = threadIdx.x;
    const float* xr = xW + (size_t)(b * DIM_L + l) * DIM_D;
    float s = 0.f;
    for (int a = 0; a < DIM_D; ++a) s += attn_v[a] * tanhf(xr[a] + hW[a]);
    sE[l] = s;
  }
  __syncthreads();

  if (threadIdx.x == 0) {
    float mx = sE[0];
    for (int l = 1; l < DIM_L; ++l) mx = fmaxf(mx, sE[l]);
    float sm = 0.f;
    for (int l = 0; l < DIM_L; ++l) { float ex = expf(sE[l] - mx); sA[l] = ex; sm += ex; }
    float inv = 1.0f / sm;
    for (int l = 0; l < DIM_L; ++l) sA[l] *= inv;
  }
  __syncthreads();

  const float* fb = feature + (size_t)(b * DIM_T + t) * DIM_R;
  bf16_t* ab = Acat + (size_t)b * KCAT;
  for (int idx = threadIdx.x; idx < DIM_R; idx += 256) {
    int l = idx >> 6;
    ab[idx] = (bf16_t)(fb[idx] * sA[l]);
  }
}

// ---------------------------------------------------------------------------
// LSTM elementwise: reduce NSPLIT gate slabs + bias, update c/h, rewrite
// the h-slice of Acat with h_new (bf16).
// ---------------------------------------------------------------------------
__global__ void __launch_bounds__(256)
lstm_update(const float* __restrict__ G,      // [NSPLIT][B][4H]
            const float* __restrict__ bias_g, // [4H]
            float* __restrict__ h, float* __restrict__ c,
            bf16_t* __restrict__ Acat)
{
  int idx = blockIdx.x * 256 + threadIdx.x;
  if (idx >= DIM_B * DIM_H) return;
  int b = idx >> 10, j = idx & (DIM_H - 1);
  const size_t slab = (size_t)DIM_B * 4 * DIM_H;
  const float* g = G + (size_t)b * 4 * DIM_H;
  float gi = bias_g[j];
  float gf = bias_g[DIM_H + j];
  float gg = bias_g[2 * DIM_H + j];
  float go = bias_g[3 * DIM_H + j];
#pragma unroll
  for (int zz = 0; zz < NSPLIT; ++zz) {
    const float* gz = g + (size_t)zz * slab;
    gi += gz[j];
    gf += gz[DIM_H + j];
    gg += gz[2 * DIM_H + j];
    go += gz[3 * DIM_H + j];
  }
  float si = 1.0f / (1.0f + expf(-gi));
  float sf = 1.0f / (1.0f + expf(-gf));
  float so = 1.0f / (1.0f + expf(-go));
  float cn = sf * c[idx] + si * tanhf(gg);
  float hn = so * tanhf(cn);
  c[idx] = cn;
  h[idx] = hn;
  Acat[(size_t)b * KCAT + DIM_R + j] = (bf16_t)hn;
}

// ---------------------------------------------------------------------------
// Head: ut[b,o] = (sum_z P[z][b,:] + Wh_b) . Wu_w[o,:] + Wu_b[o]
// Grid: 64 blocks (b). 192 threads: (o = tid/64, seg = tid%64), LDS reduce.
// ---------------------------------------------------------------------------
__global__ void __launch_bounds__(192)
head_kernel(const float* __restrict__ P,      // [NSPLIT][B][H]
            const float* __restrict__ Wh_b,
            const float* __restrict__ Wu_w, const float* __restrict__ Wu_b,
            float* __restrict__ out, int t)
{
  const int b = blockIdx.x;
  const int o = threadIdx.x >> 6;            // 0..2
  const int seg = threadIdx.x & 63;
  __shared__ float red[3][64];
  const size_t slab = (size_t)DIM_B * DIM_H;
  float s = 0.f;
  for (int m = seg * 16; m < seg * 16 + 16; ++m) {
    float pm = Wh_b[m];
#pragma unroll
    for (int zz = 0; zz < NSPLIT; ++zz) pm += P[(size_t)zz * slab + b * DIM_H + m];
    s += pm * Wu_w[(size_t)o * DIM_H + m];
  }
  red[o][seg] = s;
  __syncthreads();
  if (threadIdx.x < 3) {
    float tot = Wu_b[threadIdx.x];
    for (int k = 0; k < 64; ++k) tot += red[threadIdx.x][k];
    out[(size_t)(b * DIM_T + t) * 3 + threadIdx.x] = tot;
  }
}

// ---------------------------------------------------------------------------
// Host: workspace layout + launch sequence (graph-capture safe)
// ---------------------------------------------------------------------------
extern "C" void kernel_launch(void* const* d_in, const int* in_sizes, int n_in,
                              void* d_out, int out_size, void* d_ws, size_t ws_size,
                              hipStream_t stream)
{
  const float* feature = (const float*)d_in[0];
  const float* attn_Wx = (const float*)d_in[1];
  const float* attn_Wh = (const float*)d_in[2];
  const float* attn_v  = (const float*)d_in[3];
  const float* W_ih    = (const float*)d_in[4];
  const float* W_hh    = (const float*)d_in[5];
  const float* b_ih    = (const float*)d_in[6];
  const float* b_hh    = (const float*)d_in[7];
  const float* Wh_w    = (const float*)d_in[8];
  const float* Wh_b    = (const float*)d_in[9];
  const float* Wy_w    = (const float*)d_in[10];
  const float* Wu_w    = (const float*)d_in[11];
  const float* Wu_b    = (const float*)d_in[12];
  const float* iWh_w   = (const float*)d_in[13];
  const float* iWh_b   = (const float*)d_in[14];
  const float* iWc_w   = (const float*)d_in[15];
  const float* iWc_b   = (const float*)d_in[16];
  float* out = (float*)d_out;

  char* ws = (char*)d_ws;
  size_t off = 0;
  auto take = [&](size_t bytes) {
    size_t o = off;
    off += (bytes + 255) & ~(size_t)255;
    return o;
  };
  bf16_t* Wg    = (bf16_t*)(ws + take((size_t)4 * DIM_H * KCAT * 2)); // [4096,6784]
  bf16_t* Wo    = (bf16_t*)(ws + take((size_t)DIM_H * KCAT * 2));     // [1024,6784]
  bf16_t* featb = (bf16_t*)(ws + take((size_t)DIM_B * DIM_T * DIM_R * 2));
  bf16_t* WxT   = (bf16_t*)(ws + take((size_t)DIM_D * DIM_D * 2));
  bf16_t* Acat  = (bf16_t*)(ws + take((size_t)DIM_B * KCAT * 2));
  float*  xW    = (float*)(ws + take((size_t)DIM_B * DIM_L * DIM_D * 4));
  float*  G     = (float*)(ws + take((size_t)NSPLIT * DIM_B * 4 * DIM_H * 4));
  float*  P     = (float*)(ws + take((size_t)NSPLIT * DIM_B * DIM_H * 4));
  float*  hbuf  = (float*)(ws + take((size_t)DIM_B * DIM_H * 4));
  float*  cbuf  = (float*)(ws + take((size_t)DIM_B * DIM_H * 4));
  float*  biasg = (float*)(ws + take((size_t)4 * DIM_H * 4));
  (void)ws_size; (void)in_sizes; (void)n_in; (void)out_size;

  // --- per-call weight prep: bf16, K-concatenated, L2-resident thereafter ---
  {
    long long nWg = (long long)4 * DIM_H * KCAT;
    build_cat_bf16<<<(unsigned)((nWg + 255) / 256), 256, 0, stream>>>(
        W_ih, W_hh, Wg, 4 * DIM_H, DIM_R, DIM_H);
    long long nWo = (long long)DIM_H * KCAT;
    build_cat_bf16<<<(unsigned)((nWo + 255) / 256), 256, 0, stream>>>(
        Wy_w, Wh_w, Wo, DIM_H, DIM_R, DIM_H);
    long long nf = (long long)DIM_B * DIM_T * DIM_R;
    cvt_bf16<<<(unsigned)((nf + 255) / 256), 256, 0, stream>>>(feature, featb, nf);
    misc_init<<<16, 256, 0, stream>>>(attn_Wx, b_ih, b_hh, WxT, biasg);
  }

  // --- h0 / c0 ---
  init_state<<<DIM_B, 256, 0, stream>>>(feature, iWh_w, iWh_b, iWc_w, iWc_b,
                                        hbuf, cbuf, Acat);

  // --- 20 sequential timesteps ---
  for (int t = 0; t < DIM_T; ++t) {
    // xW = x_t @ attn_Wx   (WMMA, feature-row remap; K=64, no split)
    gemm_wmma_bf16<<<dim3(1, DIM_B * DIM_L / 64, 1), 256, 0, stream>>>(
        featb, DIM_D, WxT, DIM_D, xW, DIM_D,
        DIM_B * DIM_L, DIM_D, DIM_D, 1, t);

    // additive attention + softmax + visual -> Acat[:, 0:R]
    attn_kernel<<<DIM_B, 256, 0, stream>>>(feature, attn_Wh, attn_v, xW, hbuf,
                                           Acat, t);

    // gates = [visual, h_old] @ [W_ih | W_hh]^T  (WMMA, split-K x4 -> 128 blocks)
    gemm_wmma_bf16<<<dim3(4 * DIM_H / 128, 1, NSPLIT), 256, 0, stream>>>(
        Acat, KCAT, Wg, KCAT, G, 4 * DIM_H,
        DIM_B, 4 * DIM_H, KCAT, 0, 0);

    // LSTM update (reduces split-K slabs + bias); h-slice of Acat <- h_new
    lstm_update<<<(DIM_B * DIM_H) / 256, 256, 0, stream>>>(G, biasg, hbuf, cbuf,
                                                           Acat);

    // P = [visual, h_new] @ [Wy_w | Wh_w]^T  (WMMA, split-K x4 -> 32 blocks)
    gemm_wmma_bf16<<<dim3(DIM_H / 128, 1, NSPLIT), 256, 0, stream>>>(
        Acat, KCAT, Wo, KCAT, P, DIM_H,
        DIM_B, DIM_H, KCAT, 0, 0);

    // ut = (P_sum + Wh_b) @ Wu_w^T + Wu_b
    head_kernel<<<DIM_B, 192, 0, stream>>>(P, Wh_b, Wu_w, Wu_b, out, t);
  }
}
